// ScaledDotProductAttention_47270410059855
// MI455X (gfx1250) — matrix operand order repaired and checked by reading, not compiled, not run on hardware
//
#include <hip/hip_runtime.h>

typedef __attribute__((ext_vector_type(16))) _Float16 v16h;
typedef __attribute__((ext_vector_type(8)))  _Float16 v8h;
typedef __attribute__((ext_vector_type(4)))  _Float16 v4h;
typedef __attribute__((ext_vector_type(8)))  float    v8f;

namespace {
constexpr int kB     = 16;
constexpr int kS     = 2048;
constexpr int kD     = 128;
constexpr int kKT    = 32;            // keys per tile
constexpr int kTiles = kS / kKT;      // 64
constexpr int kKRow  = 136;           // K LDS row stride (halfs), padded: conflict-free b128 reads
constexpr int kVRow  = 40;            // V^T LDS row stride (halfs), padded
constexpr int kWaves = 4;             // 4 waves * 16 queries = 64 queries per workgroup
}

__global__ __launch_bounds__(kWaves * 32) void
fa_fwd_f16wmma(const float* __restrict__ Q, const float* __restrict__ K,
               const float* __restrict__ V, float* __restrict__ O)
{
    __shared__ alignas(16) _Float16 Ks[2][kKT * kKRow];   // K tile, row-major f16
    __shared__ alignas(16) _Float16 Vs[2][kD * kVRow];    // V tile, transposed (V^T) f16

    const int tid  = threadIdx.x;
    const int lane = tid & 31;
    const int wave = tid >> 5;
    const int b    = blockIdx.y;
    const int q0   = blockIdx.x * (kWaves * 16) + wave * 16;
    const int m16  = lane & 15;       // M-row / query-column index for this lane
    const int hi   = lane >> 4;       // which half of the wave

    // Fold 1/sqrt(D) and log2(e) into Q so softmax runs in base-2 (v_exp_f32).
    const float qscale = 0.08838834764831845f * 1.4426950408889634f;

    // ---- Q: 4 loop-invariant B-fragments (B of S^T = K * Q^T), 32 d-values each ----
    v16h qf[4];
    {
        const float* qp = Q + ((size_t)b * kS + (size_t)(q0 + m16)) * kD + hi * 16;
        #pragma unroll
        for (int c = 0; c < 4; ++c) {
            const float* p = qp + c * 32;
            float4 f0 = *(const float4*)(p + 0);
            float4 f1 = *(const float4*)(p + 4);
            float4 f2 = *(const float4*)(p + 8);
            float4 f3 = *(const float4*)(p + 12);
            v16h t;
            t[0]  = (_Float16)(f0.x * qscale); t[1]  = (_Float16)(f0.y * qscale);
            t[2]  = (_Float16)(f0.z * qscale); t[3]  = (_Float16)(f0.w * qscale);
            t[4]  = (_Float16)(f1.x * qscale); t[5]  = (_Float16)(f1.y * qscale);
            t[6]  = (_Float16)(f1.z * qscale); t[7]  = (_Float16)(f1.w * qscale);
            t[8]  = (_Float16)(f2.x * qscale); t[9]  = (_Float16)(f2.y * qscale);
            t[10] = (_Float16)(f2.z * qscale); t[11] = (_Float16)(f2.w * qscale);
            t[12] = (_Float16)(f3.x * qscale); t[13] = (_Float16)(f3.y * qscale);
            t[14] = (_Float16)(f3.z * qscale); t[15] = (_Float16)(f3.w * qscale);
            qf[c] = t;
        }
    }

    const float* Kb = K + (size_t)b * kS * kD;
    const float* Vb = V + (size_t)b * kS * kD;

    // Stage one 32-key K/V tile into LDS (all 128 threads), f32 -> f16 on the way.
    auto stage = [&](int bi, int kv0) {
        {   // K: coalesced float4 loads, row-major f16 store
            const int d = lane * 4;
            #pragma unroll
            for (int i = 0; i < 8; ++i) {
                const int kv = i * 4 + wave;
                float4 f = *(const float4*)(Kb + (size_t)(kv0 + kv) * kD + d);
                v4h h;
                h[0] = (_Float16)f.x; h[1] = (_Float16)f.y;
                h[2] = (_Float16)f.z; h[3] = (_Float16)f.w;
                *(v4h*)(&Ks[bi][kv * kKRow + d]) = h;
            }
        }
        {   // V: transpose into V^T [d][kv]; lane <-> kv row keeps store conflicts ~2-way
            const float* vr = Vb + (size_t)(kv0 + lane) * kD + wave * 32;
            #pragma unroll
            for (int j = 0; j < 8; ++j) {
                float4 f = *(const float4*)(vr + j * 4);
                const int d = wave * 32 + j * 4;
                Vs[bi][(d + 0) * kVRow + lane] = (_Float16)f.x;
                Vs[bi][(d + 1) * kVRow + lane] = (_Float16)f.y;
                Vs[bi][(d + 2) * kVRow + lane] = (_Float16)f.z;
                Vs[bi][(d + 3) * kVRow + lane] = (_Float16)f.w;
            }
        }
    };

    // A-fragment (16x32 f16): per-lane two contiguous 8-half runs (b128 LDS loads).
    auto afragK = [&](int bi, int kh, int c) -> v16h {
        const _Float16* p = &Ks[bi][(kh * 16 + m16) * kKRow + c * 32 + hi * 16];
        v8h lo = *(const v8h*)p;
        v8h hh = *(const v8h*)(p + 8);
        return __builtin_shufflevector(lo, hh, 0,1,2,3,4,5,6,7,8,9,10,11,12,13,14,15);
    };
    auto afragV = [&](int bi, int t) -> v16h {
        const _Float16* p = &Vs[bi][(t * 16 + m16) * kVRow + hi * 16];
        v8h lo = *(const v8h*)p;
        v8h hh = *(const v8h*)(p + 8);
        return __builtin_shufflevector(lo, hh, 0,1,2,3,4,5,6,7,8,9,10,11,12,13,14,15);
    };

    const v8f zero8 = {0.f,0.f,0.f,0.f,0.f,0.f,0.f,0.f};
    v8f oacc[8];                 // O^T fragments: 8 d-tiles x (16d x 16q)
    #pragma unroll
    for (int t = 0; t < 8; ++t) oacc[t] = zero8;

    float m_run = -3.402823466e38f;   // running max (base-2 domain)
    float l_run = 0.f;                // running sum

    stage(0, 0);
    __syncthreads();

    for (int kt = 0; kt < kTiles; ++kt) {
        const int bi = kt & 1;
        if (kt + 1 < kTiles) stage(bi ^ 1, (kt + 1) * kKT);

        // ---- S^T = K_tile * Q^T : two 16-key halves, K-dim = D = 4 x 32 ----
        v8f s0 = zero8, s1 = zero8;
        #pragma unroll
        for (int c = 0; c < 4; ++c) {
            s0 = __builtin_amdgcn_wmma_f32_16x16x32_f16(false, afragK(bi, 0, c),
                                                        false, qf[c], (short)0, s0, false, false);
            s1 = __builtin_amdgcn_wmma_f32_16x16x32_f16(false, afragK(bi, 1, c),
                                                        false, qf[c], (short)0, s1, false, false);
        }

        // ---- online softmax over the kv axis (per-query = per-lane column) ----
        float mx = -3.402823466e38f;
        #pragma unroll
        for (int j = 0; j < 8; ++j) mx = fmaxf(mx, fmaxf(s0[j], s1[j]));
        mx = fmaxf(mx, __shfl_xor(mx, 16, 32));
        const float m_new = fmaxf(m_run, mx);
        const float alpha = __builtin_amdgcn_exp2f(m_run - m_new);
        m_run = m_new;

        v8h ph0, ph1;
        float rs = 0.f;
        #pragma unroll
        for (int j = 0; j < 8; ++j) {
            float p0 = __builtin_amdgcn_exp2f(s0[j] - m_new);
            float p1 = __builtin_amdgcn_exp2f(s1[j] - m_new);
            rs += p0 + p1;
            ph0[j] = (_Float16)p0;
            ph1[j] = (_Float16)p1;
        }
        rs += __shfl_xor(rs, 16, 32);
        l_run = l_run * alpha + rs;

        // ---- build B operand P^T (32kv x 16q): half-wave swap of packed f16 ----
        union HU { v8h h; int i[4]; };
        HU u0, u1, w0, w1;
        u0.h = ph0; u1.h = ph1;
        #pragma unroll
        for (int k = 0; k < 4; ++k) {
            w0.i[k] = __shfl_xor(u0.i[k], 16, 32);
            w1.i[k] = __shfl_xor(u1.i[k], 16, 32);
        }
        const v8h blo = hi ? w1.h : ph0;   // kv 0-7  (lanes<16) / kv 16-23 (lanes>=16)
        const v8h bhi = hi ? ph1 : w0.h;   // kv 8-15 (lanes<16) / kv 24-31 (lanes>=16)
        const v16h pb = __builtin_shufflevector(blo, bhi,
                            0,1,2,3,4,5,6,7,8,9,10,11,12,13,14,15);

        // ---- O^T = alpha*O^T + V^T_tile * P^T : 8 d-tiles, K-dim = 32 kv ----
        #pragma unroll
        for (int t = 0; t < 8; ++t) {
            v8f o = oacc[t];
            #pragma unroll
            for (int j = 0; j < 8; ++j) o[j] *= alpha;
            oacc[t] = __builtin_amdgcn_wmma_f32_16x16x32_f16(false, afragV(bi, t),
                                                             false, pb, (short)0, o, false, false);
        }

        __syncthreads();
    }

    // ---- normalize and store: lane owns query column q0+m16, d = 16t + 8*hi + j ----
    const float inv = __builtin_amdgcn_rcpf(l_run);
    float* op = O + ((size_t)b * kS + (size_t)(q0 + m16)) * kD + hi * 8;
    #pragma unroll
    for (int t = 0; t < 8; ++t) {
        float4 a, c;
        a.x = oacc[t][0] * inv; a.y = oacc[t][1] * inv;
        a.z = oacc[t][2] * inv; a.w = oacc[t][3] * inv;
        c.x = oacc[t][4] * inv; c.y = oacc[t][5] * inv;
        c.z = oacc[t][6] * inv; c.w = oacc[t][7] * inv;
        *(float4*)(op + t * 16)     = a;
        *(float4*)(op + t * 16 + 4) = c;
    }
}

extern "C" void kernel_launch(void* const* d_in, const int* in_sizes, int n_in,
                              void* d_out, int out_size, void* d_ws, size_t ws_size,
                              hipStream_t stream) {
    const float* q = (const float*)d_in[0];
    const float* k = (const float*)d_in[1];
    const float* v = (const float*)d_in[2];
    float* o = (float*)d_out;
    (void)in_sizes; (void)n_in; (void)out_size; (void)d_ws; (void)ws_size;
    dim3 grid(kS / (kWaves * 16), kB);   // 32 x 16 = 512 workgroups
    dim3 block(kWaves * 32);             // 128 threads = 4 waves
    fa_fwd_f16wmma<<<grid, block, 0, stream>>>(q, k, v, o);
}
